// CausalSelfAttention_74448963109059
// MI455X (gfx1250) — matrix operand order repaired and checked by reading, not compiled, not run on hardware
//
#include <hip/hip_runtime.h>

// ---------------- problem constants ----------------
#define BB 2
#define TT 4096
#define CC 512
#define HH 8
#define DD 64
#define MM (BB*TT)          // 8192 rows

typedef _Float16 half_t;
typedef __attribute__((ext_vector_type(16))) _Float16 v16h;
typedef __attribute__((ext_vector_type(8)))  float    v8f;
typedef __attribute__((ext_vector_type(4)))  int      i4_t;

union V16HU { v16h v; uint4 q[2]; };
union PK4   { _Float16 h[4]; uint2 u; };

// ---------------- CDNA5 async global->LDS copy (guarded) ----------------
#if defined(__AMDGCN__) && __has_builtin(__builtin_amdgcn_global_load_async_to_lds_b128)
#define HAVE_ASYNC 1
#else
#define HAVE_ASYNC 0
#endif

#if HAVE_ASYNC
typedef __attribute__((address_space(1))) i4_t* as1_i4p;
typedef __attribute__((address_space(3))) i4_t* as3_i4p;
__device__ inline void async_copy16(const half_t* g, half_t* l) {
    __builtin_amdgcn_global_load_async_to_lds_b128(
        (as1_i4p)g, (as3_i4p)l, 0, 0);
}
#if __has_builtin(__builtin_amdgcn_s_wait_asynccnt)
#define AWAIT(n) __builtin_amdgcn_s_wait_asynccnt(n)
#else
#define AWAIT(n) asm volatile("s_wait_asynccnt %0" :: "i"(n) : "memory")
#endif
#else
// synchronous fallback (still correct under the same barrier scheme)
__device__ inline void async_copy16(const half_t* g, half_t* l) {
    *(uint4*)l = *(const uint4*)g;
}
#define AWAIT(n) ((void)0)
#endif

// ---------------- WMMA helpers ----------------
__device__ inline v8f wmma_f16(v16h a, v16h b, v8f c) {
    return __builtin_amdgcn_wmma_f32_16x16x32_f16(
        false, a, false, b, (short)0, c, false, false);
}

// A-fragment (16x32 f16): lane<16 -> row=lane, K {0..7,16..23};
// lane>=16 -> row=lane-16, K {8..15,24..31}.  ld = row stride (halves).
__device__ inline v16h load_a_frag(const half_t* base, int ld) {
    int lane = threadIdx.x & 31;
    int r = lane & 15, hi = lane >> 4;
    const half_t* p = base + r * ld + hi * 8;
    V16HU f;
    f.q[0] = *(const uint4*)(p);
    f.q[1] = *(const uint4*)(p + 16);
    return f.v;
}

// B-fragment (32x16 f16): lane&15 = column n; lanes 0-15 K {0..7,16..23},
// lanes 16-31 K {8..15,24..31}.  Tile stored [n][K] row-major, ld stride.
__device__ inline v16h load_b_frag(const half_t* base, int ld) {
    int lane = threadIdx.x & 31;
    int n = lane & 15, hi = lane >> 4;
    const half_t* p = base + n * ld + hi * 8;
    V16HU f;
    f.q[0] = *(const uint4*)(p);
    f.q[1] = *(const uint4*)(p + 16);
    return f.v;
}

// =====================================================================
// Kernel 0: fp32 -> f16 bulk convert (one float4 per thread)
// =====================================================================
__global__ __launch_bounds__(256) void cvt_kernel(
    const float* __restrict__ src, half_t* __restrict__ dst)
{
    int i = blockIdx.x * 256 + threadIdx.x;
    float4 f = ((const float4*)src)[i];
    PK4 p;
    p.h[0] = (half_t)f.x; p.h[1] = (half_t)f.y;
    p.h[2] = (half_t)f.z; p.h[3] = (half_t)f.w;
    ((uint2*)dst)[i] = p.u;
}

// ---------------- GEMM staging (f16, async) ----------------
// A tile 128x64 halves (1024 x b128, 4 per thread of 256)
__device__ inline void stage_A(const half_t* g, half_t* l, int tid) {
    #pragma unroll
    for (int i = 0; i < 4; ++i) {
        int v = tid + i * 256;
        int r = v >> 3, c = v & 7;
        async_copy16(g + (size_t)r * CC + c * 8, l + r * 64 + c * 8);
    }
}
// B tile 64x64 halves (512 x b128, 2 per thread of 256)
__device__ inline void stage_B(const half_t* g, half_t* l, int tid) {
    #pragma unroll
    for (int i = 0; i < 2; ++i) {
        int v = tid + i * 256;
        int r = v >> 3, c = v & 7;
        async_copy16(g + (size_t)r * CC + c * 8, l + r * 64 + c * 8);
    }
}

// =====================================================================
// Kernel 1: fused QKV projection, f16 in/out, K-chunk 64, double buffer.
//   mode 0 (q): scaled 1/sqrt(D), layout [B,H,T,D]
//   mode 1 (k): layout [B,H,T,D]
//   mode 2 (v): layout [B,H,D,T]   (transposed for PV B-fragments)
// Block tile 128x64, 256 threads = 8 waves (4 along M x 2 along N).
// =====================================================================
__global__ __launch_bounds__(256) void qkv_gemm_kernel(
    const half_t* __restrict__ xh,
    const half_t* __restrict__ wqh, const half_t* __restrict__ wkh,
    const half_t* __restrict__ wvh,
    half_t* __restrict__ oq, half_t* __restrict__ ok, half_t* __restrict__ ovt)
{
    __shared__ half_t As[2][128 * 64];
    __shared__ half_t Bs[2][64 * 64];

    const int mode = blockIdx.z;
    const half_t* W = (mode == 0) ? wqh : (mode == 1) ? wkh : wvh;

    const int tid  = threadIdx.x;
    const int lane = tid & 31;
    const int wave = tid >> 5;
    const int wm = wave & 3, wn = wave >> 2;
    const int m0 = blockIdx.x * 128;
    const int n0 = blockIdx.y * 64;

    const half_t* Ag = xh + (size_t)m0 * CC;
    const half_t* Bg = W  + (size_t)n0 * CC;

    v8f acc[2][2] = {};

    const int NIT = CC / 64;   // 8
    stage_A(Ag, &As[0][0], tid);
    stage_B(Bg, &Bs[0][0], tid);

    for (int it = 0; it < NIT; ++it) {
        const int buf = it & 1;
        if (it + 1 < NIT) {
            stage_A(Ag + (it + 1) * 64, &As[buf ^ 1][0], tid);
            stage_B(Bg + (it + 1) * 64, &Bs[buf ^ 1][0], tid);
            AWAIT(6);
        } else {
            AWAIT(0);
        }
        __syncthreads();

        #pragma unroll
        for (int kc = 0; kc < 64; kc += 32) {
            v16h af[2], bf[2];
            #pragma unroll
            for (int mi = 0; mi < 2; ++mi)
                af[mi] = load_a_frag(&As[buf][(wm * 32 + mi * 16) * 64 + kc], 64);
            #pragma unroll
            for (int ni = 0; ni < 2; ++ni)
                bf[ni] = load_b_frag(&Bs[buf][(wn * 32 + ni * 16) * 64 + kc], 64);
            #pragma unroll
            for (int mi = 0; mi < 2; ++mi)
                #pragma unroll
                for (int ni = 0; ni < 2; ++ni)
                    acc[mi][ni] = wmma_f16(af[mi], bf[ni], acc[mi][ni]);
        }
        __syncthreads();
    }

    const int hi  = lane >> 4;
    const int nlo = lane & 15;
    const float scale = (mode == 0) ? 0.125f : 1.0f;   // 1/sqrt(64) folded into Q
    #pragma unroll
    for (int mi = 0; mi < 2; ++mi)
        #pragma unroll
        for (int ni = 0; ni < 2; ++ni)
            #pragma unroll
            for (int j = 0; j < 8; ++j) {
                int row = m0 + wm * 32 + mi * 16 + j + 8 * hi;  // b*T+t
                int col = n0 + wn * 32 + ni * 16 + nlo;         // 0..511
                int b = row >> 12, t = row & 4095;
                int h = col >> 6,  d = col & 63;
                half_t val = (half_t)(acc[mi][ni][j] * scale);
                if (mode == 0)
                    oq[(((size_t)(b * HH + h)) * TT + t) * DD + d] = val;
                else if (mode == 1)
                    ok[(((size_t)(b * HH + h)) * TT + t) * DD + d] = val;
                else
                    ovt[(((size_t)(b * HH + h)) * DD + d) * TT + t] = val;
            }
}

// =====================================================================
// Kernel 2: causal flash attention, 64-key tiles, double-buffered async
// staging.  4 waves x 16 Q rows; online fp32 softmax; P via per-wave LDS.
// =====================================================================
__device__ inline void stage_K(const half_t* g, half_t* l, int tid) {
    #pragma unroll
    for (int i = 0; i < 4; ++i) {          // 64x64 halves, 512 b128 / 128 thr
        int v = tid + i * 128;
        int r = v >> 3, c = v & 7;
        async_copy16(g + (size_t)r * DD + c * 8, l + r * 64 + c * 8);
    }
}
__device__ inline void stage_V(const half_t* g, half_t* l, int tid) {
    #pragma unroll
    for (int i = 0; i < 4; ++i) {          // 64 d-rows x 64 keys
        int v = tid + i * 128;
        int r = v >> 3, c = v & 7;
        async_copy16(g + (size_t)r * TT + c * 8, l + r * 64 + c * 8);
    }
}

__global__ __launch_bounds__(128) void attn_kernel(
    const half_t* __restrict__ q, const half_t* __restrict__ k,
    const half_t* __restrict__ vt, half_t* __restrict__ attn)
{
    __shared__ half_t Ks[2][64 * 64];
    __shared__ half_t Vs[2][64 * 64];
    __shared__ half_t Ps[4][16 * 64];

    const int tid  = threadIdx.x;
    const int lane = tid & 31;
    const int wave = tid >> 5;
    const int hi   = lane >> 4;
    const int nlo  = lane & 15;

    const int nq = TT / 64;
    const int bh = blockIdx.x / nq;
    const int qt = blockIdx.x % nq;
    const int b  = bh >> 3;
    const int h  = bh & 7;
    const int q0 = qt * 64;

    const half_t* qb = q  + (size_t)bh * TT * DD;
    const half_t* kb = k  + (size_t)bh * TT * DD;
    const half_t* vb = vt + (size_t)bh * DD * TT;

    const int rowbase = q0 + wave * 16;
    v16h aq0 = load_a_frag(qb + (size_t)rowbase * DD + 0,  DD);
    v16h aq1 = load_a_frag(qb + (size_t)rowbase * DD + 32, DD);

    v8f   o[4] = {};
    float m_[8], l_[8];
    #pragma unroll
    for (int j = 0; j < 8; ++j) { m_[j] = -1.0e30f; l_[j] = 0.0f; }

    const int nkt = qt + 1;                  // 64-key tiles up to diagonal

    stage_K(kb, &Ks[0][0], tid);
    stage_V(vb, &Vs[0][0], tid);

    for (int kt = 0; kt < nkt; ++kt) {
        const int kb0 = kt * 64;
        const int buf = kt & 1;
        if (kt + 1 < nkt) {
            stage_K(kb + (size_t)(kb0 + 64) * DD, &Ks[buf ^ 1][0], tid);
            stage_V(vb + (kb0 + 64),              &Vs[buf ^ 1][0], tid);
            AWAIT(8);
        } else {
            AWAIT(0);
        }
        __syncthreads();

        // ---- S = Q*K^T : four 16x16 C tiles across 64 keys ----
        v8f s[4];
        #pragma unroll
        for (int nt = 0; nt < 4; ++nt) {
            v16h bk0 = load_b_frag(&Ks[buf][(nt * 16) * 64 + 0],  64);
            v16h bk1 = load_b_frag(&Ks[buf][(nt * 16) * 64 + 32], 64);
            v8f t = {};
            t = wmma_f16(aq0, bk0, t);
            t = wmma_f16(aq1, bk1, t);
            s[nt] = t;
        }

        // ---- causal mask: only the diagonal tile ----
        if (kt == nkt - 1) {
            #pragma unroll
            for (int nt = 0; nt < 4; ++nt) {
                int col = kb0 + nt * 16 + nlo;
                #pragma unroll
                for (int j = 0; j < 8; ++j) {
                    int row = q0 + wave * 16 + j + 8 * hi;
                    if (col > row) s[nt][j] = -3.0e38f;
                }
            }
        }

        // ---- online softmax per row ----
        #pragma unroll
        for (int j = 0; j < 8; ++j) {
            float v0 = s[0][j], v1 = s[1][j], v2 = s[2][j], v3 = s[3][j];
            float mv = fmaxf(fmaxf(v0, v1), fmaxf(v2, v3));
            #pragma unroll
            for (int off = 1; off < 16; off <<= 1)
                mv = fmaxf(mv, __shfl_xor(mv, off, 32));
            float mn  = fmaxf(m_[j], mv);
            float psc = __expf(m_[j] - mn);
            m_[j] = mn;
            float p0 = __expf(v0 - mn), p1 = __expf(v1 - mn);
            float p2 = __expf(v2 - mn), p3 = __expf(v3 - mn);
            float rs = (p0 + p1) + (p2 + p3);
            #pragma unroll
            for (int off = 1; off < 16; off <<= 1)
                rs += __shfl_xor(rs, off, 32);
            l_[j] = l_[j] * psc + rs;
            s[0][j] = p0; s[1][j] = p1; s[2][j] = p2; s[3][j] = p3;
            #pragma unroll
            for (int n = 0; n < 4; ++n)
                o[n][j] = o[n][j] * psc;
        }

        // ---- P -> LDS (wave-private; per-wave DS ops are in-order) ----
        half_t* P = &Ps[wave][0];
        #pragma unroll
        for (int nt = 0; nt < 4; ++nt)
            #pragma unroll
            for (int j = 0; j < 8; ++j)
                P[(j + 8 * hi) * 64 + nt * 16 + nlo] = (half_t)s[nt][j];

        // ---- O += P * V ----
        v16h ap0 = load_a_frag(P + 0,  64);
        v16h ap1 = load_a_frag(P + 32, 64);
        #pragma unroll
        for (int n = 0; n < 4; ++n) {
            v16h bv0 = load_b_frag(&Vs[buf][(n * 16) * 64 + 0],  64);
            v16h bv1 = load_b_frag(&Vs[buf][(n * 16) * 64 + 32], 64);
            o[n] = wmma_f16(ap0, bv0, o[n]);
            o[n] = wmma_f16(ap1, bv1, o[n]);
        }
        __syncthreads();   // all waves done with buf before it is restaged
    }

    // ---- normalize, store f16 into [B,T,C] ----
    #pragma unroll
    for (int j = 0; j < 8; ++j) {
        float inv = 1.0f / l_[j];
        int t = q0 + wave * 16 + j + 8 * hi;
        #pragma unroll
        for (int n = 0; n < 4; ++n) {
            int d = n * 16 + nlo;
            attn[((size_t)(b * TT + t)) * CC + h * DD + d] =
                (half_t)(o[n][j] * inv);
        }
    }
}

// =====================================================================
// Kernel 3: out projection  out = attn(f16) @ Wout^T -> fp32
// =====================================================================
__global__ __launch_bounds__(256) void out_gemm_kernel(
    const half_t* __restrict__ a, const half_t* __restrict__ wh,
    float* __restrict__ out)
{
    __shared__ half_t As[2][128 * 64];
    __shared__ half_t Bs[2][64 * 64];

    const int tid  = threadIdx.x;
    const int lane = tid & 31;
    const int wave = tid >> 5;
    const int wm = wave & 3, wn = wave >> 2;
    const int m0 = blockIdx.x * 128;
    const int n0 = blockIdx.y * 64;

    const half_t* Ag = a  + (size_t)m0 * CC;
    const half_t* Bg = wh + (size_t)n0 * CC;

    v8f acc[2][2] = {};

    const int NIT = CC / 64;
    stage_A(Ag, &As[0][0], tid);
    stage_B(Bg, &Bs[0][0], tid);

    for (int it = 0; it < NIT; ++it) {
        const int buf = it & 1;
        if (it + 1 < NIT) {
            stage_A(Ag + (it + 1) * 64, &As[buf ^ 1][0], tid);
            stage_B(Bg + (it + 1) * 64, &Bs[buf ^ 1][0], tid);
            AWAIT(6);
        } else {
            AWAIT(0);
        }
        __syncthreads();

        #pragma unroll
        for (int kc = 0; kc < 64; kc += 32) {
            v16h af[2], bf[2];
            #pragma unroll
            for (int mi = 0; mi < 2; ++mi)
                af[mi] = load_a_frag(&As[buf][(wm * 32 + mi * 16) * 64 + kc], 64);
            #pragma unroll
            for (int ni = 0; ni < 2; ++ni)
                bf[ni] = load_b_frag(&Bs[buf][(wn * 32 + ni * 16) * 64 + kc], 64);
            #pragma unroll
            for (int mi = 0; mi < 2; ++mi)
                #pragma unroll
                for (int ni = 0; ni < 2; ++ni)
                    acc[mi][ni] = wmma_f16(af[mi], bf[ni], acc[mi][ni]);
        }
        __syncthreads();
    }

    const int hi  = lane >> 4;
    const int nlo = lane & 15;
    #pragma unroll
    for (int mi = 0; mi < 2; ++mi)
        #pragma unroll
        for (int ni = 0; ni < 2; ++ni)
            #pragma unroll
            for (int j = 0; j < 8; ++j) {
                int row = m0 + wm * 32 + mi * 16 + j + 8 * hi;
                int col = n0 + wn * 32 + ni * 16 + nlo;
                out[(size_t)row * CC + col] = acc[mi][ni][j];
            }
}

// =====================================================================
extern "C" void kernel_launch(void* const* d_in, const int* in_sizes, int n_in,
                              void* d_out, int out_size, void* d_ws, size_t ws_size,
                              hipStream_t stream) {
    (void)in_sizes; (void)n_in; (void)out_size; (void)ws_size;
    const float* x    = (const float*)d_in[0];
    const float* Wq   = (const float*)d_in[1];
    const float* Wk   = (const float*)d_in[2];
    const float* Wv   = (const float*)d_in[3];
    const float* Wout = (const float*)d_in[4];

    // workspace layout (halves):
    //   xh   : MM*CC      (8 MB)   -- reused later as attn output
    //   wqh/wkh/wvh/woh : CC*CC each (0.5 MB each)
    //   q,k,vt : MM*CC each (8 MB each)       total ~34 MB
    half_t* xh   = (half_t*)d_ws;
    half_t* wqh  = xh  + (size_t)MM * CC;
    half_t* wkh  = wqh + (size_t)CC * CC;
    half_t* wvh  = wkh + (size_t)CC * CC;
    half_t* woh  = wvh + (size_t)CC * CC;
    half_t* q    = woh + (size_t)CC * CC;
    half_t* k    = q   + (size_t)MM * CC;
    half_t* vt   = k   + (size_t)MM * CC;
    half_t* attn = xh;   // alias: xh is dead after the QKV GEMM

    cvt_kernel<<<MM * CC / 1024, 256, 0, stream>>>(x, xh);
    cvt_kernel<<<CC * CC / 1024, 256, 0, stream>>>(Wq, wqh);
    cvt_kernel<<<CC * CC / 1024, 256, 0, stream>>>(Wk, wkh);
    cvt_kernel<<<CC * CC / 1024, 256, 0, stream>>>(Wv, wvh);
    cvt_kernel<<<CC * CC / 1024, 256, 0, stream>>>(Wout, woh);

    dim3 g1(MM / 128, CC / 64, 3);
    qkv_gemm_kernel<<<g1, 256, 0, stream>>>(xh, wqh, wkh, wvh, q, k, vt);

    attn_kernel<<<BB * HH * (TT / 64), 128, 0, stream>>>(q, k, vt, attn);

    dim3 g3(MM / 128, CC / 64);
    out_gemm_kernel<<<g3, 256, 0, stream>>>(attn, woh, (float*)d_out);
}
